// CrossSharedUnit_2465311228652
// MI455X (gfx1250) — compile-verified, matrix operand order checked by reading, not run on hardware
//
#include <hip/hip_runtime.h>
#include <hip/hip_bf16.h>

// Problem sizes (fixed by the reference).
#define B_ 32
#define S_ 128
#define H_ 512
#define K_ 8

typedef __attribute__((ext_vector_type(16))) __bf16        v16bf;
typedef __attribute__((ext_vector_type(8)))  float         v8f;
typedef __attribute__((ext_vector_type(4)))  unsigned int  u32x4;
typedef __attribute__((ext_vector_type(4)))  int           i32x4;
typedef __attribute__((ext_vector_type(4)))  float         f32x4;

// ---------------------------------------------------------------------------
// CDNA5 feature guards (device pass only; fall back cleanly elsewhere).
// ---------------------------------------------------------------------------
#if defined(__gfx1250__) && __has_builtin(__builtin_amdgcn_global_load_async_to_lds_b128)
#define USE_ASYNC_LDS 1
#else
#define USE_ASYNC_LDS 0
#endif

#if USE_ASYNC_LDS
typedef __attribute__((address_space(1))) i32x4 gbl_i32x4;
typedef __attribute__((address_space(3))) i32x4 lds_i32x4;
#endif

// 16-byte global -> LDS copy. Async DMA (ASYNCcnt-tracked) when available.
__device__ __forceinline__ void copy16_to_lds(void* lds_dst, const void* gsrc) {
#if USE_ASYNC_LDS
  __builtin_amdgcn_global_load_async_to_lds_b128(
      (gbl_i32x4*)(void*)gsrc, (lds_i32x4*)lds_dst, 0, 0);
#else
  *(u32x4*)lds_dst = *(const u32x4*)gsrc;
#endif
}

__device__ __forceinline__ void async_lds_fence() {
#if USE_ASYNC_LDS
#if __has_builtin(__builtin_amdgcn_s_wait_asynccnt)
  __builtin_amdgcn_s_wait_asynccnt(0);
#else
  asm volatile("s_wait_asynccnt 0" ::: "memory");
#endif
#endif
}

// Native v_tanh_f32 (TRANS pipe, co-executes with the WMMA pipe) if exposed.
__device__ __forceinline__ float fast_tanh(float x) {
#if defined(__gfx1250__) && __has_builtin(__builtin_amdgcn_tanhf)
  return __builtin_amdgcn_tanhf(x);
#else
  return tanhf(x);
#endif
}

__device__ __forceinline__ unsigned short f32_to_bf16(float f) {
  unsigned int u = __float_as_uint(f);
  u += 0x7fffu + ((u >> 16) & 1u);        // round-to-nearest-even
  return (unsigned short)(u >> 16);
}

__device__ __forceinline__ v8f v8f_zero() {
  v8f z;
  #pragma unroll
  for (int i = 0; i < 8; ++i) z[i] = 0.0f;
  return z;
}

// Load a 16x32 bf16 WMMA fragment from row-major [n][k] (or [m][k]) storage.
// Per CDNA5 ISA: lanes 0-15 hold K 0..7 / 16..23, lanes 16-31 hold K 8..15 / 24..31.
// Two 16B chunks per lane -> ds_load_b128 x2.
__device__ __forceinline__ v16bf load_frag_bf16(const unsigned short* base,
                                                int row0, int k0, int stride) {
  const int lane = threadIdx.x & 31;
  const int r    = lane & 15;
  const int koff = (lane >> 4) << 3;      // 0 or 8
  const unsigned short* p = base + (size_t)(row0 + r) * stride + k0 + koff;
  union { u32x4 u[2]; v16bf v; } cvt;
  cvt.u[0] = *(const u32x4*)(p);
  cvt.u[1] = *(const u32x4*)(p + 16);
  return cvt.v;
}

// ---------------------------------------------------------------------------
// Stage 0: fp32 -> bf16 convert (hiddens)
// ---------------------------------------------------------------------------
__global__ __launch_bounds__(256) void convert_bf16_kernel(
    const float* __restrict__ src, unsigned short* __restrict__ dst, int n) {
  int i = blockIdx.x * blockDim.x + threadIdx.x;
  if (i < n) dst[i] = f32_to_bf16(src[i]);
}

// Stage 0b: Gt[k][g][h] = bf16(G[h][k][g])  (makes the H contraction contiguous)
__global__ __launch_bounds__(256) void transpose_g_kernel(
    const float* __restrict__ G, unsigned short* __restrict__ Gt) {
  int i = blockIdx.x * blockDim.x + threadIdx.x;   // over K*H*H, write-coalesced
  if (i < K_ * H_ * H_) {
    int h = i % H_;
    int g = (i / H_) % H_;
    int k = i / (H_ * H_);
    Gt[i] = f32_to_bf16(G[((size_t)h * K_ + k) * H_ + g]);
  }
}

// ---------------------------------------------------------------------------
// Stage 1: fused bilinear score.
// One block (8 waves, 256 thr) per (batch, branch). Wave w owns score rows
// [16w,16w+16). LDS: left 128KB | right 128KB | G-slice 32KB | M scratch 8KB.
// score[s,t] = sum_k v[k] * tanh( sum_g (sum_h left[s,h] Gk[h,g]) right[t,g] )
// ---------------------------------------------------------------------------
__global__ __launch_bounds__(256) void bilinear_score_kernel(
    const unsigned short* __restrict__ aspect_bf,
    const unsigned short* __restrict__ pol_bf,
    const unsigned short* __restrict__ Gt_ap,
    const unsigned short* __restrict__ Gt_pa,
    const float* __restrict__ v_a, const float* __restrict__ v_p,
    float* __restrict__ score_a, float* __restrict__ score_p) {
  extern __shared__ unsigned short smemB[];
  const int bb     = blockIdx.x;
  const int branch = blockIdx.y;

  // Branch 0: left=aspect, right=polarity, G=G_ap. Branch 1 (faithful to ref):
  // left=aspect, right=aspect, G=G_pa.
  const unsigned short* left_g  = aspect_bf + (size_t)bb * S_ * H_;
  const unsigned short* right_g = ((branch == 0) ? pol_bf : aspect_bf) + (size_t)bb * S_ * H_;
  const unsigned short* Gt      = (branch == 0) ? Gt_ap : Gt_pa;
  const float*          vk      = (branch == 0) ? v_a : v_p;
  float*                score   = ((branch == 0) ? score_a : score_p) + (size_t)bb * S_ * S_;

  unsigned short* lds_left  = smemB;                   // [128][512] bf16
  unsigned short* lds_right = lds_left + S_ * H_;      // [128][512] bf16
  unsigned short* lds_G     = lds_right + S_ * H_;     // [32][512]  bf16 (g-slice, h contiguous)
  unsigned short* lds_M     = lds_G + 32 * H_;         // [128][32]  bf16 (per-wave GEMM1 out)

  // Resident tiles for the whole block: 256KB of the 320KB WGP LDS.
  for (int i = threadIdx.x; i < (S_ * H_) / 8; i += 256) {
    copy16_to_lds(lds_left + i * 8, left_g + i * 8);
    copy16_to_lds(lds_right + i * 8, right_g + i * 8);
  }
  async_lds_fence();
  __syncthreads();

  const int w    = threadIdx.x >> 5;
  const int lane = threadIdx.x & 31;
  const int nn   = lane & 15;
  const int mb   = (lane >> 4) << 3;

  v8f score_acc[8];
  #pragma unroll
  for (int t = 0; t < 8; ++t) score_acc[t] = v8f_zero();

  for (int k = 0; k < K_; ++k) {
    v8f raw_acc[8];
    #pragma unroll
    for (int t = 0; t < 8; ++t) raw_acc[t] = v8f_zero();

    for (int gb = 0; gb < H_ / 32; ++gb) {
      // --- stage contiguous 32KB G slice: Gt[k][gb*32 .. gb*32+32][0..H) ---
      __syncthreads();  // previous iteration's consumers of lds_G are done
      {
        const unsigned short* gs = Gt + ((size_t)k * H_ + (size_t)gb * 32) * H_;
        for (int i = threadIdx.x; i < (32 * H_) / 8; i += 256)
          copy16_to_lds(lds_G + i * 8, gs + i * 8);
      }
      if (gb + 1 < H_ / 32) {  // prefetch next slice into cache hierarchy
        __builtin_prefetch(Gt + ((size_t)k * H_ + (size_t)(gb + 1) * 32) * H_
                              + (size_t)threadIdx.x * 64, 0, 1);
      }
      async_lds_fence();
      __syncthreads();

      // --- GEMM1: M[16x32] = left[wave rows, :] x Gk[:, gb cols], K=H=512 ---
      v8f mc0 = v8f_zero(), mc1 = v8f_zero();
      for (int hb = 0; hb < H_ / 32; ++hb) {
        v16bf a  = load_frag_bf16(lds_left, w * 16, hb * 32, H_);
        v16bf g0 = load_frag_bf16(lds_G, 0,  hb * 32, H_);
        v16bf g1 = load_frag_bf16(lds_G, 16, hb * 32, H_);
        mc0 = __builtin_amdgcn_wmma_f32_16x16x32_bf16(false, a, false, g0, (short)0, mc0, false, false);
        mc1 = __builtin_amdgcn_wmma_f32_16x16x32_bf16(false, a, false, g1, (short)0, mc1, false, false);
      }
      // C-layout -> row-major bf16 in per-wave LDS scratch (wave-local, no barrier)
      #pragma unroll
      for (int r = 0; r < 8; ++r) {
        lds_M[(w * 16 + mb + r) * 32 + nn]      = f32_to_bf16(mc0[r]);
        lds_M[(w * 16 + mb + r) * 32 + 16 + nn] = f32_to_bf16(mc1[r]);
      }

      // --- GEMM2: raw[wave rows, all t] += M x right^T, K=32 (this g block) ---
      v16bf a2 = load_frag_bf16(lds_M, w * 16, 0, 32);
      #pragma unroll
      for (int tt = 0; tt < 8; ++tt) {
        v16bf bt = load_frag_bf16(lds_right, tt * 16, gb * 32, H_);
        raw_acc[tt] = __builtin_amdgcn_wmma_f32_16x16x32_bf16(false, a2, false, bt, (short)0, raw_acc[tt], false, false);
      }
    }

    // score += v[k] * tanh(raw)  (native v_tanh_f32 on the TRANS pipe)
    const float vkk = vk[k];
    #pragma unroll
    for (int tt = 0; tt < 8; ++tt)
      #pragma unroll
      for (int e = 0; e < 8; ++e)
        score_acc[tt][e] += vkk * fast_tanh(raw_acc[tt][e]);
  }

  // write score[s,t] (C layout: lane half selects M 0..7 vs 8..15)
  #pragma unroll
  for (int tt = 0; tt < 8; ++tt)
    #pragma unroll
    for (int r = 0; r < 8; ++r)
      score[(size_t)(w * 16 + mb + r) * S_ + tt * 16 + nn] = score_acc[tt][r];
}

// ---------------------------------------------------------------------------
// Stage 2: softmax over t + out = resid + attn @ other_hidden (WMMA bf16).
// LDS: score f32 64KB | attn bf16 32KB | other^T bf16 128KB  = 224KB.
// other is staged TRANSPOSED ([h][t]) so B-fragments are clean b128 loads.
// ---------------------------------------------------------------------------
__global__ __launch_bounds__(256) void attn_out_kernel(
    const float* __restrict__ score_a, const float* __restrict__ score_p,
    const unsigned short* __restrict__ aspect_bf,
    const unsigned short* __restrict__ pol_bf,
    const float* __restrict__ aspect_f32, const float* __restrict__ pol_f32,
    float* __restrict__ out) {
  extern __shared__ float smemD[];
  const int bb     = blockIdx.x;
  const int branch = blockIdx.y;

  const float* score           = ((branch == 0) ? score_a : score_p) + (size_t)bb * S_ * S_;
  const unsigned short* other  = ((branch == 0) ? pol_bf : aspect_bf) + (size_t)bb * S_ * H_;
  const float* resid           = ((branch == 0) ? aspect_f32 : pol_f32) + (size_t)bb * S_ * H_;
  float* outp = out + (size_t)branch * B_ * S_ * H_ + (size_t)bb * S_ * H_;

  float*          lds_s    = smemD;                                  // [128][128] f32
  unsigned short* lds_attn = (unsigned short*)(lds_s + S_ * S_);     // [128][128] bf16
  unsigned short* lds_othT = lds_attn + S_ * S_;                     // [512][128] bf16 (h-major)

  // score tile: async DMA straight into LDS
  for (int i = threadIdx.x; i < (S_ * S_) / 4; i += 256)
    copy16_to_lds(lds_s + i * 4, score + i * 4);
  // other hidden: coalesced global read, transposed LDS write
  for (int i = threadIdx.x; i < (S_ * H_) / 8; i += 256) {
    const int t  = i >> 6;                 // H_/8 == 64 chunks per row
    const int h0 = (i & 63) << 3;
    u32x4 d = ((const u32x4*)other)[i];
    const unsigned short* ds = (const unsigned short*)&d;
    #pragma unroll
    for (int e = 0; e < 8; ++e) lds_othT[(size_t)(h0 + e) * S_ + t] = ds[e];
  }
  async_lds_fence();
  __syncthreads();

  // softmax over t: one thread per row s
  if (threadIdx.x < S_) {
    float* row = lds_s + (size_t)threadIdx.x * S_;
    float m = row[0];
    for (int t = 1; t < S_; ++t) m = fmaxf(m, row[t]);
    float sum = 0.0f;
    for (int t = 0; t < S_; ++t) { float e = __expf(row[t] - m); row[t] = e; sum += e; }
    const float inv = 1.0f / sum;
    unsigned short* arow = lds_attn + (size_t)threadIdx.x * S_;
    for (int t = 0; t < S_; ++t) arow[t] = f32_to_bf16(row[t] * inv);
  }
  __syncthreads();

  const int w    = threadIdx.x >> 5;
  const int lane = threadIdx.x & 31;
  const int nn   = lane & 15;
  const int mb   = (lane >> 4) << 3;

  v16bf afr[4];
  #pragma unroll
  for (int kb = 0; kb < 4; ++kb)
    afr[kb] = load_frag_bf16(lds_attn, w * 16, kb * 32, S_);

  for (int ct = 0; ct < H_ / 16; ++ct) {
    v8f acc = v8f_zero();
    #pragma unroll
    for (int kb = 0; kb < 4; ++kb) {
      v16bf bf = load_frag_bf16(lds_othT, ct * 16, kb * 32, S_);
      acc = __builtin_amdgcn_wmma_f32_16x16x32_bf16(false, afr[kb], false, bf, (short)0, acc, false, false);
    }
    #pragma unroll
    for (int r = 0; r < 8; ++r) {
      const int s = w * 16 + mb + r;
      const int h = ct * 16 + nn;
      outp[(size_t)s * H_ + h] = acc[r] + resid[(size_t)s * H_ + h];
    }
  }
}

// ---------------------------------------------------------------------------
extern "C" void kernel_launch(void* const* d_in, const int* in_sizes, int n_in,
                              void* d_out, int out_size, void* d_ws, size_t ws_size,
                              hipStream_t stream) {
  const float* aspect   = (const float*)d_in[0];
  const float* polarity = (const float*)d_in[1];
  const float* G_ap     = (const float*)d_in[2];
  const float* G_pa     = (const float*)d_in[3];
  const float* v_a      = (const float*)d_in[4];
  const float* v_p      = (const float*)d_in[5];
  float* out = (float*)d_out;

  // Workspace layout (20 MB total):
  unsigned short* aspect_bf = (unsigned short*)d_ws;                    // 4 MB
  unsigned short* pol_bf    = aspect_bf + (size_t)B_ * S_ * H_;         // 4 MB
  unsigned short* Gt_ap     = pol_bf + (size_t)B_ * S_ * H_;            // 4 MB
  unsigned short* Gt_pa     = Gt_ap + (size_t)H_ * K_ * H_;             // 4 MB
  float* score_a = (float*)(Gt_pa + (size_t)H_ * K_ * H_);              // 2 MB
  float* score_p = score_a + (size_t)B_ * S_ * S_;                      // 2 MB

  const int nconv = B_ * S_ * H_;
  convert_bf16_kernel<<<(nconv + 255) / 256, 256, 0, stream>>>(aspect, aspect_bf, nconv);
  convert_bf16_kernel<<<(nconv + 255) / 256, 256, 0, stream>>>(polarity, pol_bf, nconv);
  const int ng = K_ * H_ * H_;
  transpose_g_kernel<<<(ng + 255) / 256, 256, 0, stream>>>(G_ap, Gt_ap);
  transpose_g_kernel<<<(ng + 255) / 256, 256, 0, stream>>>(G_pa, Gt_pa);

  dim3 grid(B_, 2);
  const size_t ldsB = (size_t)(S_ * H_ + S_ * H_ + 32 * H_ + S_ * 32) * sizeof(unsigned short); // 303104
  bilinear_score_kernel<<<grid, 256, ldsB, stream>>>(
      aspect_bf, pol_bf, Gt_ap, Gt_pa, v_a, v_p, score_a, score_p);

  const size_t ldsD = (size_t)S_ * S_ * 4 + (size_t)S_ * S_ * 2 + (size_t)S_ * H_ * 2; // 229376
  attn_out_kernel<<<grid, 256, ldsD, stream>>>(
      score_a, score_p, aspect_bf, pol_bf, aspect, polarity, out);
}
